// QuantizedExpert_47047071760973
// MI455X (gfx1250) — matrix operand order repaired and checked by reading, not compiled, not run on hardware
//
#include <hip/hip_runtime.h>
#include <stdint.h>

typedef __attribute__((ext_vector_type(16))) _Float16 v16h;
typedef __attribute__((ext_vector_type(8)))  _Float16 v8h;
typedef __attribute__((ext_vector_type(4)))  _Float16 v4h;
typedef __attribute__((ext_vector_type(2)))  _Float16 h2;
typedef __attribute__((ext_vector_type(8)))  float    v8f;
typedef __attribute__((ext_vector_type(4)))  float    v4f;

#define BM 256
#define BN 128
#define BK 64
// LDS row stride in halves: 72 halves = 144 B -> 16B-aligned rows, bank skew
#define LSTR 72
#define A_HALVES (BM * LSTR)
#define B_HALVES (BN * LSTR)
#define SMEM_BYTES ((2 * A_HALVES + 2 * B_HALVES) * 2)

// ---- gfx1250 async-copy helpers (ASYNCcnt path) ----
__device__ __forceinline__ void async_load_b128(uint32_t lds_byte_off, const void* g) {
  asm volatile("global_load_async_to_lds_b128 %0, %1, off"
               :: "v"(lds_byte_off), "v"((uint64_t)(uintptr_t)g)
               : "memory");
}
__device__ __forceinline__ void wait_async0() {
  asm volatile("s_wait_asynccnt 0x0" ::: "memory");
}

// x fp32 -> f16 (one-time, bandwidth-trivial)
__global__ __launch_bounds__(256) void f32_to_f16(const float* __restrict__ in,
                                                  _Float16* __restrict__ out) {
  size_t i = ((size_t)blockIdx.x * 256 + threadIdx.x) * 8;
  v4f a = *(const v4f*)(in + i);
  v4f b = *(const v4f*)(in + i + 4);
  v4h ha, hb;
  ha[0] = (_Float16)a.x; ha[1] = (_Float16)a.y; ha[2] = (_Float16)a.z; ha[3] = (_Float16)a.w;
  hb[0] = (_Float16)b.x; hb[1] = (_Float16)b.y; hb[2] = (_Float16)b.z; hb[3] = (_Float16)b.w;
  *(v4h*)(out + i)     = ha;
  *(v4h*)(out + i + 4) = hb;
}

// D = A(f16) * W^T (+bias, opt relu). W int4-packed, per-row f16 scale.
template <bool RELU, bool OUT_HALF>
__global__ __launch_bounds__(256)
void qgemm_wmma(const _Float16* __restrict__ Ah,     // [M][K] f16
                const int8_t* __restrict__ Wp,       // [N][K/2] packed
                const _Float16* __restrict__ scales, // [N]
                const float* __restrict__ bias,      // [N]
                void* __restrict__ Out,              // [M][N]
                int M, int N, int K) {
  extern __shared__ _Float16 smem[];   // [2*A | 2*B], dynamic => LDS offset 0

  const int t    = threadIdx.x;
  const int wid  = t >> 5;
  const int lane = t & 31;
  const int bm0  = blockIdx.y * BM;
  const int bn0  = blockIdx.x * BN;

  // wave grid 4 (M) x 2 (N): each wave 64x64 = 4x4 WMMA tiles
  const int wm = (wid >> 1) * 64;
  const int wn = (wid & 1) * 64;

  v8f acc[4][4] = {};

  // ---- B staging (dequant) mapping: thread -> (row, 16-byte chunk) ----
  const int wrow  = t >> 1;
  const int wbyte = (t & 1) * 16;
  const int Kb    = K >> 1;
  const int8_t* wrow_ptr = Wp + (size_t)(bn0 + wrow) * Kb + wbyte;
  const uint32_t bOff = (uint32_t)(wrow * LSTR + wbyte * 2);  // halves, within B buf

  // magic dequant constants: as_f16(0x6400|(u^8)) = 1024 + (u^8);  val = that - 1032
  const _Float16 sh = scales[bn0 + wrow];
  const h2 s2 = {sh, sh};
  const _Float16 ch = (_Float16)(-1032.0f * (float)sh);
  const h2 c2 = {ch, ch};

  // ---- A staging (async DMA): 8 x b128 per thread (256 rows x 64 halves) ----
  auto issueA = [&](int k0, int p) {
    const uint32_t base = (uint32_t)(p * A_HALVES);
#pragma unroll
    for (int q = 0; q < 8; q++) {
      int c    = t + q * 256;          // chunk id 0..2047
      int row  = c >> 3;
      int koff = (c & 7) * 8;
      const _Float16* g = Ah + (size_t)(bm0 + row) * K + k0 + koff;
      async_load_b128((base + (uint32_t)(row * LSTR + koff)) * 2u, g);
    }
  };

  // dequant 16 packed bytes -> 32 f16 (perm + pk_fma, no int<->float cvt)
  auto dequant = [&](int k0, v8h* w) {
    const int8_t* p8 = wrow_ptr + (k0 >> 1);
    uint64_t bits01 = *(const uint64_t*)(p8);
    uint64_t bits23 = *(const uint64_t*)(p8 + 8);
    uint32_t words[4] = {(uint32_t)bits01, (uint32_t)(bits01 >> 32),
                         (uint32_t)bits23, (uint32_t)(bits23 >> 32)};
#pragma unroll
    for (int q = 0; q < 4; q++) {
      uint32_t wd = words[q];
      uint32_t e = ((wd >> 4) & 0x0F0F0F0Fu) ^ 0x08080808u;  // even cols (hi nibble)
      uint32_t o = (wd & 0x0F0F0F0Fu) ^ 0x08080808u;         // odd cols (lo nibble)
#pragma unroll
      for (int i = 0; i < 4; i++) {
        // result bytes: [e.byte_i, 0x00, o.byte_i, 0x00] ; sel 12 => 0x00
        uint32_t sel = 0x0C000C00u | ((uint32_t)i << 16) | (uint32_t)(4 + i);
        uint32_t pck = __builtin_amdgcn_perm(e, o, sel) | 0x64006400u;
        h2 hv = __builtin_bit_cast(h2, pck);
        h2 r  = hv * s2 + c2;                                // v_pk_fma_f16
        w[q][2 * i]     = r[0];
        w[q][2 * i + 1] = r[1];
      }
    }
  };
  auto storeB = [&](int p, const v8h* w) {
    _Float16* dst = smem + 2 * A_HALVES + p * B_HALVES + bOff;
    *(v8h*)(dst)      = w[0];
    *(v8h*)(dst + 8)  = w[1];
    *(v8h*)(dst + 16) = w[2];
    *(v8h*)(dst + 24) = w[3];
  };

  // ---- prologue: stage k=0 into buffer 0 ----
  issueA(0, 0);
  {
    v8h w[4];
    dequant(0, w);
    storeB(0, w);
  }
  wait_async0();
  __syncthreads();

  const int steps = K / BK;
  const int mA  = lane & 15;
  const int kbA = (lane < 16) ? 0 : 16;
  const int nB  = lane & 15;
  const int kbB = (lane < 16) ? 0 : 16;

  for (int k = 0; k < steps; ++k) {
    const int cur = k & 1, nxt = cur ^ 1;
    const bool more = (k + 1 < steps);
    v8h w[4];
    if (more) {
      issueA((k + 1) * BK, nxt);       // DMA next slab, overlaps compute
      dequant((k + 1) * BK, w);        // W global load issued early; VALU co-execs
    }

    const _Float16* Abuf = smem + cur * A_HALVES;
    const _Float16* Bbuf = smem + 2 * A_HALVES + cur * B_HALVES;

#pragma unroll
    for (int kk = 0; kk < BK; kk += 32) {
      // B fragments: lane col n=lane&15; lanes 0-15 K=kk+0..15, 16-31 K=kk+16..31
      v16h bfrag[4];
#pragma unroll
      for (int j = 0; j < 4; j++) {
        const _Float16* base = Bbuf + (wn + j * 16 + nB) * LSTR + kk + kbB;
        v8h lo = *(const v8h*)(base);
        v8h hi = *(const v8h*)(base + 8);
#pragma unroll
        for (int e = 0; e < 8; e++) { bfrag[j][e] = lo[e]; bfrag[j][e + 8] = hi[e]; }
      }
      // A fragments: lanes 0-15: M=lane, K 0..15+kk; lanes 16-31: K 16..31+kk
#pragma unroll
      for (int i = 0; i < 4; i++) {
        const _Float16* base = Abuf + (wm + i * 16 + mA) * LSTR + kk;
        v8h lo = *(const v8h*)(base + kbA);
        v8h hi = *(const v8h*)(base + kbA + 8);
        v16h a;
#pragma unroll
        for (int e = 0; e < 8; e++) { a[e] = lo[e]; a[e + 8] = hi[e]; }
#pragma unroll
        for (int j = 0; j < 4; j++)
          acc[i][j] = __builtin_amdgcn_wmma_f32_16x16x32_f16(
              false, a, false, bfrag[j], (short)0, acc[i][j], false, false);
      }
    }

    if (more) storeB(nxt, w);
    wait_async0();
    __syncthreads();
  }

  // ---- epilogue: bias (+relu). C/D: VGPR v -> M=v+(lane<16?0:8), N=lane&15
  const int mOut = (lane < 16) ? 0 : 8;
  const int nOut = lane & 15;
#pragma unroll
  for (int j = 0; j < 4; j++) {
    const int gn = bn0 + wn + j * 16 + nOut;
    const float bv = bias[gn];
#pragma unroll
    for (int i = 0; i < 4; i++) {
#pragma unroll
      for (int v = 0; v < 8; v++) {
        const int gm = bm0 + wm + i * 16 + mOut + v;
        float r = acc[i][j][v] + bv;
        if (RELU) r = r > 0.0f ? r : 0.0f;
        if (OUT_HALF)
          ((_Float16*)Out)[(size_t)gm * N + gn] = (_Float16)r;
        else
          ((float*)Out)[(size_t)gm * N + gn] = r;
      }
    }
  }
}

extern "C" void kernel_launch(void* const* d_in, const int* in_sizes, int n_in,
                              void* d_out, int out_size, void* d_ws, size_t ws_size,
                              hipStream_t stream) {
  const float*    x   = (const float*)d_in[0];
  const int8_t*   w1p = (const int8_t*)d_in[1];
  const _Float16* s1  = (const _Float16*)d_in[2];
  const float*    b1  = (const float*)d_in[3];
  const int8_t*   w2p = (const int8_t*)d_in[4];
  const _Float16* s2  = (const _Float16*)d_in[5];
  const float*    b2  = (const float*)d_in[6];

  const int d    = in_sizes[6];      // 4096
  const int H    = in_sizes[3];      // 8192
  const int Ntok = in_sizes[0] / d;  // 4096

  float*    out = (float*)d_out;
  _Float16* xh  = (_Float16*)d_ws;               // [Ntok][d] f16
  _Float16* h   = xh + (size_t)Ntok * d;         // [Ntok][H] f16

  size_t nx = (size_t)Ntok * d;
  f32_to_f16<<<dim3((unsigned)(nx / (256 * 8))), dim3(256), 0, stream>>>(x, xh);

  qgemm_wmma<true, true><<<dim3(H / BN, Ntok / BM), dim3(256), SMEM_BYTES, stream>>>(
      xh, w1p, s1, b1, (void*)h, Ntok, H, d);

  qgemm_wmma<false, false><<<dim3(d / BN, Ntok / BM), dim3(256), SMEM_BYTES, stream>>>(
      h, w2p, s2, b2, (void*)out, Ntok, d, H);
}
